// ViT_66958540145107
// MI455X (gfx1250) — compile-verified, matrix operand order checked by reading, not run on hardware
//
#include <hip/hip_runtime.h>
#include <hip/hip_bf16.h>
#include <math.h>

// ---------------------------------------------------------------------------
// ViT-Base forward for MI455X (gfx1250, wave32, WMMA).
// Precision: bf16 operands, f32 accumulation via v_wmma_f32_16x16x32_bf16.
// GEMM inner loop is software-pipelined: tile t+1 global loads are in flight
// while tile t WMMAs execute from LDS.
// ---------------------------------------------------------------------------

typedef __bf16 bf16_t;
typedef bf16_t v16bf __attribute__((ext_vector_type(16)));
typedef float  v8f   __attribute__((ext_vector_type(8)));

#define EMB   768
#define SEQ   197
#define SP    208          // SEQ padded to 13*16 for WMMA tiling
#define BATCH 64
#define NHEAD 12
#define HD    64
#define NBLK  12
#define FF    3072
#define MTOK  (BATCH * SEQ)     // 12608
#define MPAT  (BATCH * 196)     // 12544

#define TILE_M 128
#define TILE_N 64
#define TILE_K 32
// LDS row strides: multiples of 8 elements (16B) so b128 LDS stores are
// legally aligned; r*stride mod 64-banks hits 16 distinct banks for r=0..15.
#define LDSS   40
#define ATS    72

union U4B8 { uint4 u; bf16_t h[8]; };

// ---------------------------------------------------------------------------
// Generic batched bf16 GEMM:  C = op( A[M,K] * B[K,N] + bias )
// modes: 0 = f32 store, 1 = f32 accumulate (+=), 2 = bf16 GELU store,
//        3 = bf16 store.
// Batch z: inner = z % innerN, outer = z / innerN; offsets via per-level strides.
// ---------------------------------------------------------------------------
__global__ __launch_bounds__(256) void gemm_bf16_wmma(
    const bf16_t* __restrict__ A, int lda, long long sAin, long long sAout,
    const bf16_t* __restrict__ B, int ldb, long long sBin, long long sBout,
    float* __restrict__ Cf, bf16_t* __restrict__ Cb, int ldc,
    long long sCin, long long sCout,
    const float* __restrict__ bias, long long sBiasIn, long long sBiasOut,
    int M, int N, int Kpad, int Kvalid, int innerN, int mode)
{
    __shared__ __align__(16) bf16_t As[TILE_M * LDSS];
    __shared__ __align__(16) bf16_t Bs[TILE_N * LDSS];

    const int tid  = threadIdx.x;
    const int wave = tid >> 5;
    const int lane = tid & 31;
    const int half = lane >> 4;
    const int lm   = lane & 15;
    const int m0   = blockIdx.x * TILE_M;
    const int n0   = blockIdx.y * TILE_N;
    const int z    = blockIdx.z;
    const int zi   = z % innerN;
    const int zo   = z / innerN;

    const bf16_t* Ab = A + (long long)zi * sAin + (long long)zo * sAout;
    const bf16_t* Bb = B + (long long)zi * sBin + (long long)zo * sBout;
    const long long cOff = (long long)zi * sCin + (long long)zo * sCout;
    const long long bOff = (long long)zi * sBiasIn + (long long)zo * sBiasOut;

    const bool interior = (m0 + TILE_M <= M) && (n0 + TILE_N <= N);
    const int  kFull    = interior ? (Kvalid / TILE_K) : 0;  // branch-free steps

    // Fast-path per-thread staging coordinates.
    const int ar  = tid >> 2;           // A row (0..63; second row is +64)
    const int ac  = (tid & 3) * 8;      // A col chunk (8 bf16)
    const int bk_ = tid >> 3;           // B k-row (0..31)
    const int bn8 = (tid & 7) * 8;      // B n chunk (8 bf16)

    // K-pair base offsets for 16-bit A/B fragments (ISA 7.12.2):
    // v0..3 -> K = half*8 + 2v ; v4..7 -> K = 16 + half*8 + 2(v-4)
    int kb[8];
#pragma unroll
    for (int v = 0; v < 4; ++v) { kb[v] = half * 8 + 2 * v; kb[v + 4] = 16 + half * 8 + 2 * v; }

    v8f acc[4];
#pragma unroll
    for (int t = 0; t < 4; ++t) acc[t] = (v8f){};

    const int arow = wave * 16 + lm;
    auto computeStep = [&]() {
        v16bf afrag;
#pragma unroll
        for (int v = 0; v < 8; ++v) {
            afrag[2 * v]     = As[arow * LDSS + kb[v]];
            afrag[2 * v + 1] = As[arow * LDSS + kb[v] + 1];
        }
#pragma unroll
        for (int t = 0; t < 4; ++t) {
            v16bf bfrag;
            const int bn = t * 16 + lm;
#pragma unroll
            for (int v = 0; v < 8; ++v) {
                bfrag[2 * v]     = Bs[bn * LDSS + kb[v]];
                bfrag[2 * v + 1] = Bs[bn * LDSS + kb[v] + 1];
            }
            acc[t] = __builtin_amdgcn_wmma_f32_16x16x32_bf16(
                false, afrag, false, bfrag, (short)0, acc[t], false, false);
        }
    };

    // ---- software-pipelined interior K-loop -------------------------------
    if (kFull > 0) {
        const bf16_t* pA0 = Ab + (long long)(m0 + ar) * lda + ac;
        const bf16_t* pA1 = pA0 + (long long)64 * lda;
        const bf16_t* pB  = Bb + (long long)bk_ * ldb + n0 + bn8;
        const long long bStep = (long long)TILE_K * ldb;

        uint4 aR0 = *(const uint4*)pA0;
        uint4 aR1 = *(const uint4*)pA1;
        uint4 bR  = *(const uint4*)pB;

        for (int s = 0; s < kFull; ++s) {
            // L2 prefetch two tiles ahead (speculative: dropped if invalid).
            __builtin_prefetch(pA0 + 2 * TILE_K, 0, 1);
            __builtin_prefetch(pB + 2 * bStep, 0, 1);

            // Stage registers -> LDS (b128 stores; B transposed to n-major).
            *(uint4*)&As[ar * LDSS + ac]        = aR0;
            *(uint4*)&As[(ar + 64) * LDSS + ac] = aR1;
            {
                U4B8 t; t.u = bR;
#pragma unroll
                for (int u = 0; u < 8; ++u) Bs[(bn8 + u) * LDSS + bk_] = t.h[u];
            }
            __syncthreads();

            // Kick off next tile's global loads before computing this tile.
            pA0 += TILE_K; pA1 += TILE_K; pB += bStep;
            if (s + 1 < kFull) {
                aR0 = *(const uint4*)pA0;
                aR1 = *(const uint4*)pA1;
                bR  = *(const uint4*)pB;
            }

            computeStep();
            __syncthreads();
        }
    }

    // ---- guarded tail (edge blocks and K remainder/padding) ---------------
    for (int kt = kFull * TILE_K; kt < Kpad; kt += TILE_K) {
#pragma unroll
        for (int j = 0; j < (TILE_M * TILE_K) / 256; ++j) {
            const int e = tid + j * 256;
            const int r = e >> 5, kk = e & 31;
            const int gm = m0 + r, gk = kt + kk;
            bf16_t v = (bf16_t)0.f;
            if (gm < M && gk < Kvalid) v = Ab[(long long)gm * lda + gk];
            As[r * LDSS + kk] = v;
        }
#pragma unroll
        for (int j = 0; j < (TILE_N * TILE_K) / 256; ++j) {
            const int e = tid + j * 256;
            const int k = e >> 6, n = e & 63;
            const int gk = kt + k, gn = n0 + n;
            bf16_t v = (bf16_t)0.f;
            if (gk < Kvalid && gn < N) v = Bb[(long long)gk * ldb + gn];
            Bs[n * LDSS + k] = v;
        }
        __syncthreads();
        computeStep();
        __syncthreads();
    }

    // Epilogue. C/D layout: VGPR r -> M = r + 8*half, N = lane%16.
#pragma unroll
    for (int t = 0; t < 4; ++t) {
        const int gn = n0 + t * 16 + lm;
        if (gn >= N) continue;
        const float bv = bias ? bias[bOff + gn] : 0.f;
#pragma unroll
        for (int r = 0; r < 8; ++r) {
            const int gm = m0 + wave * 16 + r + 8 * half;
            if (gm >= M) continue;
            float val = acc[t][r] + bv;
            const long long ci = cOff + (long long)gm * ldc + gn;
            if (mode == 0)      Cf[ci] = val;
            else if (mode == 1) Cf[ci] += val;
            else if (mode == 2) Cb[ci] = (bf16_t)(0.5f * val * (1.f + erff(val * 0.70710678118f)));
            else                Cb[ci] = (bf16_t)val;
        }
    }
}

// ---------------------------------------------------------------------------
// Attention scores: S[z,q,k] = scale * Q_bh[q,:] . K_bh[k,:]
// K rows are contiguous in the head dim, so B fragments read contiguously.
// ---------------------------------------------------------------------------
__global__ __launch_bounds__(256) void attn_scores_wmma(
    const bf16_t* __restrict__ Q, const bf16_t* __restrict__ K,
    float* __restrict__ S, float scale)
{
    __shared__ __align__(16) bf16_t Qs[64 * ATS];
    __shared__ __align__(16) bf16_t Ks[64 * ATS];

    const int tid  = threadIdx.x;
    const int wave = tid >> 5;
    const int lane = tid & 31;
    const int half = lane >> 4;
    const int lm   = lane & 15;
    const int z = blockIdx.z;                 // b*NHEAD + h
    const int b = z / NHEAD, h = z % NHEAD;
    const int q0 = blockIdx.x * 64, k0 = blockIdx.y * 64;

    const bf16_t* Qb = Q + (long long)b * SEQ * EMB + h * HD;
    const bf16_t* Kb = K + (long long)b * SEQ * EMB + h * HD;

    // 64 rows x 64 cols, 8-element chunks: 512 chunks, 2 per thread.
#pragma unroll
    for (int j = 0; j < 2; ++j) {
        const int e = tid + j * 256;
        const int r = e >> 3, c8 = (e & 7) * 8;
        uint4 qv = {0, 0, 0, 0}, kv = {0, 0, 0, 0};
        if (q0 + r < SEQ) qv = *(const uint4*)(Qb + (long long)(q0 + r) * EMB + c8);
        if (k0 + r < SEQ) kv = *(const uint4*)(Kb + (long long)(k0 + r) * EMB + c8);
        *(uint4*)&Qs[r * ATS + c8] = qv;
        *(uint4*)&Ks[r * ATS + c8] = kv;
    }
    __syncthreads();

    const int wm = wave >> 1, wn = wave & 1;  // 4 M-tiles x 2 N-halves
    int kb[8];
#pragma unroll
    for (int v = 0; v < 4; ++v) { kb[v] = half * 8 + 2 * v; kb[v + 4] = 16 + half * 8 + 2 * v; }

    v8f acc[2];
    acc[0] = (v8f){}; acc[1] = (v8f){};

#pragma unroll
    for (int kt = 0; kt < HD; kt += 32) {
        v16bf a;
        const int arow = wm * 16 + lm;
#pragma unroll
        for (int v = 0; v < 8; ++v) {
            a[2 * v]     = Qs[arow * ATS + kt + kb[v]];
            a[2 * v + 1] = Qs[arow * ATS + kt + kb[v] + 1];
        }
#pragma unroll
        for (int t = 0; t < 2; ++t) {
            v16bf bf_;
            const int bn = wn * 32 + t * 16 + lm;
#pragma unroll
            for (int v = 0; v < 8; ++v) {
                bf_[2 * v]     = Ks[bn * ATS + kt + kb[v]];
                bf_[2 * v + 1] = Ks[bn * ATS + kt + kb[v] + 1];
            }
            acc[t] = __builtin_amdgcn_wmma_f32_16x16x32_bf16(
                false, a, false, bf_, (short)0, acc[t], false, false);
        }
    }

#pragma unroll
    for (int t = 0; t < 2; ++t) {
        const int gk = k0 + wn * 32 + t * 16 + lm;
        if (gk >= SEQ) continue;
#pragma unroll
        for (int r = 0; r < 8; ++r) {
            const int gq = q0 + wm * 16 + r + 8 * half;
            if (gq >= SEQ) continue;
            S[(long long)z * SEQ * SP + (long long)gq * SP + gk] = acc[t][r] * scale;
        }
    }
}

// ---------------------------------------------------------------------------
// Row softmax over 197 logits -> bf16 probs, zero-padded to 208 columns.
// ---------------------------------------------------------------------------
__global__ __launch_bounds__(256) void softmax_rows(
    const float* __restrict__ S, bf16_t* __restrict__ P)
{
    __shared__ float red[256];
    const int tid = threadIdx.x;
    const long long row = blockIdx.x;          // z*SEQ + q
    const long long z = row / SEQ;
    const int q = (int)(row % SEQ);
    const float* s = S + z * (SEQ * SP) + (long long)q * SP;

    float v = (tid < SEQ) ? s[tid] : -1e30f;
    red[tid] = v; __syncthreads();
    for (int off = 128; off > 0; off >>= 1) {
        if (tid < off) red[tid] = fmaxf(red[tid], red[tid + off]);
        __syncthreads();
    }
    const float mx = red[0]; __syncthreads();
    float e = (tid < SEQ) ? __expf(v - mx) : 0.f;
    red[tid] = e; __syncthreads();
    for (int off = 128; off > 0; off >>= 1) {
        if (tid < off) red[tid] += red[tid + off];
        __syncthreads();
    }
    const float inv = 1.f / red[0];
    if (tid < SP)
        P[z * (SEQ * SP) + (long long)q * SP + tid] = (bf16_t)((tid < SEQ) ? e * inv : 0.f);
}

// ---------------------------------------------------------------------------
// LayerNorm over 768 features; fp32 in (residual), bf16 out (GEMM operand).
// ---------------------------------------------------------------------------
__global__ __launch_bounds__(256) void layernorm_rows(
    const float* __restrict__ X, const float* __restrict__ g,
    const float* __restrict__ b, bf16_t* __restrict__ out)
{
    __shared__ float s1[256], s2[256];
    const int tid = threadIdx.x;
    const long long row = blockIdx.x;
    const float* x = X + row * EMB;
    const float v0 = x[tid], v1 = x[tid + 256], v2 = x[tid + 512];
    s1[tid] = v0 + v1 + v2;
    s2[tid] = v0 * v0 + v1 * v1 + v2 * v2;
    __syncthreads();
    for (int off = 128; off > 0; off >>= 1) {
        if (tid < off) { s1[tid] += s1[tid + off]; s2[tid] += s2[tid + off]; }
        __syncthreads();
    }
    const float mean = s1[0] * (1.f / EMB);
    const float var  = s2[0] * (1.f / EMB) - mean * mean;
    const float rstd = rsqrtf(var + 1e-5f);
    bf16_t* o = out + row * EMB;
    o[tid]       = (bf16_t)((v0 - mean) * rstd * g[tid]       + b[tid]);
    o[tid + 256] = (bf16_t)((v1 - mean) * rstd * g[tid + 256] + b[tid + 256]);
    o[tid + 512] = (bf16_t)((v2 - mean) * rstd * g[tid + 512] + b[tid + 512]);
}

// ---------------------------------------------------------------------------
// images (B,3,224,224) -> patch matrix Xp[B*196, 768] (bf16), feature =
// c*256 + iy*16 + ix  (matches reference transpose(0,2,4,1,3,5)).
// ---------------------------------------------------------------------------
__global__ __launch_bounds__(256) void patchify(
    const float* __restrict__ img, bf16_t* __restrict__ Xp)
{
    const long long i = (long long)blockIdx.x * 256 + threadIdx.x;
    if (i >= (long long)MPAT * EMB) return;
    const int f = (int)(i % EMB);
    const long long row = i / EMB;
    const int p = (int)(row % 196);
    const int b = (int)(row / 196);
    const int c = f >> 8, rem = f & 255, iy = rem >> 4, ix = rem & 15;
    const int py = p / 14, px = p % 14;
    Xp[i] = (bf16_t)img[(((long long)b * 3 + c) * 224 + (py * 16 + iy)) * 224 + (px * 16 + ix)];
}

// ---------------------------------------------------------------------------
// X[b,s,:] = (s==0 ? cls : tokens[b,s-1]) + pos[s]
// ---------------------------------------------------------------------------
__global__ __launch_bounds__(256) void assemble_seq(
    const float* __restrict__ tokens, const float* __restrict__ cls,
    const float* __restrict__ pos, float* __restrict__ X)
{
    const long long i = (long long)blockIdx.x * 256 + threadIdx.x;
    if (i >= (long long)MTOK * EMB) return;
    const int c = (int)(i % EMB);
    const long long r = i / EMB;
    const int s = (int)(r % SEQ);
    const long long b = r / SEQ;
    const float t = (s == 0) ? cls[c] : tokens[(b * 196 + (s - 1)) * EMB + c];
    X[i] = t + pos[(long long)s * EMB + c];
}

// ---------------------------------------------------------------------------
// Classifier head + softmax on CLS token (negligible FLOPs -> VALU kernel).
// ---------------------------------------------------------------------------
__global__ void head_softmax(
    const float* __restrict__ X, const float* __restrict__ Wh,
    const float* __restrict__ bh, float* __restrict__ out)
{
    const int b = blockIdx.x;
    const int tid = threadIdx.x;
    __shared__ float logits[10];
    if (tid < 10) {
        float acc = bh[tid];
        const float* x = X + (long long)b * SEQ * EMB;   // CLS row
        for (int k = 0; k < EMB; ++k) acc += x[k] * Wh[k * 10 + tid];
        logits[tid] = acc;
    }
    __syncthreads();
    if (tid < 10) {
        float m = -1e30f;
        for (int j = 0; j < 10; ++j) m = fmaxf(m, logits[j]);
        float s = 0.f;
        for (int j = 0; j < 10; ++j) s += __expf(logits[j] - m);
        out[b * 10 + tid] = __expf(logits[tid] - m) / s;
    }
}

__global__ __launch_bounds__(256) void f32_to_bf16(
    const float* __restrict__ in, bf16_t* __restrict__ out, long long n)
{
    const long long i = (long long)blockIdx.x * 256 + threadIdx.x;
    if (i < n) out[i] = (bf16_t)in[i];
}

// ---------------------------------------------------------------------------
extern "C" void kernel_launch(void* const* d_in, const int* in_sizes, int n_in,
                              void* d_out, int out_size, void* d_ws, size_t ws_size,
                              hipStream_t stream)
{
    (void)in_sizes; (void)n_in; (void)out_size; (void)ws_size;

    const float* images = (const float*)d_in[0];
    const float* Wmap   = (const float*)d_in[1];
    const float* bmap   = (const float*)d_in[2];
    const float* cls    = (const float*)d_in[3];
    const float* pos    = (const float*)d_in[4];
    const float* ln1_g  = (const float*)d_in[5];
    const float* ln1_b  = (const float*)d_in[6];
    const float* Wq     = (const float*)d_in[7];
    const float* bq     = (const float*)d_in[8];
    const float* Wk     = (const float*)d_in[9];
    const float* bk     = (const float*)d_in[10];
    const float* Wv     = (const float*)d_in[11];
    const float* bv     = (const float*)d_in[12];
    const float* ln2_g  = (const float*)d_in[13];
    const float* ln2_b  = (const float*)d_in[14];
    const float* W1     = (const float*)d_in[15];
    const float* b1     = (const float*)d_in[16];
    const float* W2     = (const float*)d_in[17];
    const float* b2     = (const float*)d_in[18];
    const float* Whead  = (const float*)d_in[19];
    const float* bhead  = (const float*)d_in[20];

    char* ws = (char*)d_ws;
    size_t o = 0;
    auto alloc = [&](size_t bytes) -> char* {
        o = (o + 255) & ~(size_t)255;
        char* p = ws + o;
        o += bytes;
        return p;
    };

    // Workspace layout (~316 MB total):
    bf16_t* WmapB = (bf16_t*)alloc((size_t)EMB * EMB * 2);                 // 1.18 MB
    bf16_t* WqB   = (bf16_t*)alloc((size_t)NHEAD * HD * HD * 2);           // 98 KB (per-block scratch)
    bf16_t* WkB   = (bf16_t*)alloc((size_t)NHEAD * HD * HD * 2);
    bf16_t* WvB   = (bf16_t*)alloc((size_t)NHEAD * HD * HD * 2);
    bf16_t* W1B   = (bf16_t*)alloc((size_t)EMB * FF * 2);                  // 4.7 MB
    bf16_t* W2B   = (bf16_t*)alloc((size_t)FF * EMB * 2);                  // 4.7 MB
    float*  X     = (float*)alloc((size_t)MTOK * EMB * 4);                 // 38.7 MB residual
    bf16_t* Hb    = (bf16_t*)alloc((size_t)MTOK * EMB * 2);                // 19.4 MB (LN out / patch matrix)
    bf16_t* Qb    = (bf16_t*)alloc((size_t)MTOK * EMB * 2);
    bf16_t* Kb    = (bf16_t*)alloc((size_t)MTOK * EMB * 2);
    bf16_t* Vb    = (bf16_t*)alloc((size_t)MTOK * EMB * 2);
    char*   regA  = alloc((size_t)BATCH * NHEAD * SEQ * SP * 4);           // 125.9 MB: scores f32 / MLP hidden bf16
    char*   regB  = alloc((size_t)BATCH * NHEAD * SEQ * SP * 2);           // 62.9 MB: probs bf16 / patch tokens f32

    const float scale = 0.28867513459481287f;  // 1/sqrt(H) per reference
    const long long sAttn = (long long)SEQ * SP;   // 40976 per (b,h)

    // ---- Patch embedding --------------------------------------------------
    f32_to_bf16<<<(EMB * EMB + 255) / 256, 256, 0, stream>>>(Wmap, WmapB, (long long)EMB * EMB);
    patchify<<<((long long)MPAT * EMB + 255) / 256, 256, 0, stream>>>(images, Hb);
    gemm_bf16_wmma<<<dim3(MPAT / TILE_M, EMB / TILE_N, 1), 256, 0, stream>>>(
        Hb, EMB, 0, 0, WmapB, EMB, 0, 0,
        (float*)regB, nullptr, EMB, 0, 0,
        bmap, 0, 0, MPAT, EMB, EMB, EMB, 1, /*mode=*/0);
    assemble_seq<<<((long long)MTOK * EMB + 255) / 256, 256, 0, stream>>>(
        (const float*)regB, cls, pos, X);

    const int gmM = (MTOK + TILE_M - 1) / TILE_M;   // 99

    // ---- Transformer blocks ----------------------------------------------
    for (int i = 0; i < NBLK; ++i) {
        const long long wqkvN = (long long)NHEAD * HD * HD;  // 49152
        f32_to_bf16<<<(wqkvN + 255) / 256, 256, 0, stream>>>(Wq + (long long)i * wqkvN, WqB, wqkvN);
        f32_to_bf16<<<(wqkvN + 255) / 256, 256, 0, stream>>>(Wk + (long long)i * wqkvN, WkB, wqkvN);
        f32_to_bf16<<<(wqkvN + 255) / 256, 256, 0, stream>>>(Wv + (long long)i * wqkvN, WvB, wqkvN);
        const long long w1N = (long long)EMB * FF;
        f32_to_bf16<<<(w1N + 255) / 256, 256, 0, stream>>>(W1 + (long long)i * w1N, W1B, w1N);
        f32_to_bf16<<<(w1N + 255) / 256, 256, 0, stream>>>(W2 + (long long)i * w1N, W2B, w1N);

        // LN1 -> bf16
        layernorm_rows<<<MTOK, 256, 0, stream>>>(X, ln1_g + i * EMB, ln1_b + i * EMB, Hb);

        // QKV: 12 batched per-head 64x64 GEMMs each, bf16 outputs (mode 3)
        gemm_bf16_wmma<<<dim3(gmM, 1, NHEAD), 256, 0, stream>>>(
            Hb, EMB, HD, 0, WqB, HD, (long long)HD * HD, 0,
            nullptr, Qb, EMB, HD, 0,
            bq + i * EMB, HD, 0, MTOK, HD, HD, HD, NHEAD, /*mode=*/3);
        gemm_bf16_wmma<<<dim3(gmM, 1, NHEAD), 256, 0, stream>>>(
            Hb, EMB, HD, 0, WkB, HD, (long long)HD * HD, 0,
            nullptr, Kb, EMB, HD, 0,
            bk + i * EMB, HD, 0, MTOK, HD, HD, HD, NHEAD, /*mode=*/3);
        gemm_bf16_wmma<<<dim3(gmM, 1, NHEAD), 256, 0, stream>>>(
            Hb, EMB, HD, 0, WvB, HD, (long long)HD * HD, 0,
            nullptr, Vb, EMB, HD, 0,
            bv + i * EMB, HD, 0, MTOK, HD, HD, HD, NHEAD, /*mode=*/3);

        // Scores (Q K^T * scale) and softmax
        attn_scores_wmma<<<dim3(4, 4, BATCH * NHEAD), 256, 0, stream>>>(
            Qb, Kb, (float*)regA, scale);
        softmax_rows<<<BATCH * NHEAD * SEQ, 256, 0, stream>>>(
            (const float*)regA, (bf16_t*)regB);

        // attn @ V, residual-accumulated into X (mode 1)
        gemm_bf16_wmma<<<dim3(2, 1, BATCH * NHEAD), 256, 0, stream>>>(
            (const bf16_t*)regB, SP, sAttn, sAttn * NHEAD,
            Vb, EMB, HD, (long long)SEQ * EMB,
            X, nullptr, EMB, HD, (long long)SEQ * EMB,
            nullptr, 0, 0, SEQ, HD, SP, SEQ, NHEAD, /*mode=*/1);

        // LN2 -> bf16
        layernorm_rows<<<MTOK, 256, 0, stream>>>(X, ln2_g + i * EMB, ln2_b + i * EMB, Hb);

        // MLP: GEMM1 + exact GELU -> bf16 hidden (mode 2)
        gemm_bf16_wmma<<<dim3(gmM, FF / TILE_N, 1), 256, 0, stream>>>(
            Hb, EMB, 0, 0, W1B, FF, 0, 0,
            nullptr, (bf16_t*)regA, FF, 0, 0,
            b1 + i * FF, 0, 0, MTOK, FF, EMB, EMB, 1, /*mode=*/2);
        // GEMM2, residual-accumulated into X (mode 1)
        gemm_bf16_wmma<<<dim3(gmM, EMB / TILE_N, 1), 256, 0, stream>>>(
            (const bf16_t*)regA, FF, 0, 0, W2B, EMB, 0, 0,
            X, nullptr, EMB, 0, 0,
            b2 + i * EMB, 0, 0, MTOK, EMB, FF, FF, 1, /*mode=*/1);
    }

    // ---- Head -------------------------------------------------------------
    head_softmax<<<BATCH, 32, 0, stream>>>(X, Whead, bhead, (float*)d_out);
}
